// InterMA_88356067213479
// MI455X (gfx1250) — matrix operand order repaired and checked by reading, not compiled, not run on hardware
//
#include <hip/hip_runtime.h>

// Problem dims (match reference)
#define NN 4096      // nodes
#define NE 2048      // edges
#define DIN 256      // input dim
#define NH 8         // heads
#define DH 64        // head dim
#define HD 512       // NH*DH
#define MAXREL 10

// ---------------- CDNA5 WMMA helpers (wave32, 16x16x32 bf16 -> f32) ----------------
typedef __attribute__((ext_vector_type(16))) __bf16 v16bf;
typedef __attribute__((ext_vector_type(8)))  float  v8f;

union V16U { v16bf v; unsigned short u[16]; uint4 q[2]; };

__device__ __forceinline__ unsigned short f2bf(float f) {
  unsigned int x = __float_as_uint(f);
  x += 0x7FFFu + ((x >> 16) & 1u);       // round-to-nearest-even
  return (unsigned short)(x >> 16);
}

// Swizzled tiles: each lane's 16 operand halves are contiguous (32B) in LDS.
// A 16x32 (MxK): lane = m + 16*(k>=16),     t = k&15  [t-runs are k-contiguous]
// B 32x16 (KxN): lane = n + 16*(k>=16),     t = k&15              [t-runs are k-contiguous]
__device__ __forceinline__ v16bf load_sw(const unsigned short* s, int lane) {
  V16U r;
  const uint4* p = (const uint4*)(s + (lane << 4));
  r.q[0] = p[0];
  r.q[1] = p[1];
  return r.v;
}

__device__ __forceinline__ v8f wmma_bf16(v16bf a, v16bf b, v8f c) {
  return __builtin_amdgcn_wmma_f32_16x16x32_bf16(false, a, false, b, (short)0, c, false, false);
}

// CDNA5 async global->LDS DMA (ASYNCcnt-tracked), 16B per lane.
// LDS byte address = low 32 bits of the flat shared-memory pointer (ISA 10.2).
__device__ __forceinline__ void async_ld_b128(void* lds_ptr, const void* g) {
  unsigned int l = (unsigned int)(size_t)lds_ptr;
  unsigned long long ga = (unsigned long long)(size_t)g;
  asm volatile("global_load_async_to_lds_b128 %0, %1, off" :: "v"(l), "v"(ga) : "memory");
}
__device__ __forceinline__ void wait_async0() {
  asm volatile("s_wait_asynccnt 0x0" ::: "memory");
}

// ---------------- one-time: weights f32 -> bf16, pre-swizzled into B-tile layout ----------------
// out tile id = (ty*(K/32) + kb)*(N/16) + ns ; within tile: lane*16 + t  (B layout)
__global__ void cvt_swz_kernel(const float* __restrict__ in, unsigned short* __restrict__ out,
                               int K, int N, int total8) {
  int idx = blockIdx.x * 256 + threadIdx.x;
  if (idx >= total8) return;
  int o = idx * 8;
  int tile = o >> 9, off = o & 511;
  int lane = off >> 4, t0 = off & 15;      // t0 in {0,8}
  int ntiles = N >> 4, ktiles = K >> 5;
  int ns = tile % ntiles;
  int kb = (tile / ntiles) % ktiles;
  int ty = tile / (ntiles * ktiles);
  int n = ns * 16 + (lane & 15);
  int kbase = kb * 32 + (lane >> 4) * 16 + t0;
  const float* src = in + ((size_t)ty * K + kbase) * N + n;
  unsigned int w[4];
#pragma unroll
  for (int u = 0; u < 4; ++u) {
    float lo = src[(size_t)(2 * u) * N];
    float hi = src[(size_t)(2 * u + 1) * N];
    w[u] = (unsigned int)f2bf(lo) | ((unsigned int)f2bf(hi) << 16);
  }
  uint4 q; q.x = w[0]; q.y = w[1]; q.z = w[2]; q.w = w[3];
  *(uint4*)(out + o) = q;
}

// ---------------- LayerNorm -> bf16 ----------------
__global__ void ln_bf16_kernel(const float* __restrict__ x, const float* __restrict__ g,
                               const float* __restrict__ b, unsigned short* __restrict__ out) {
  __shared__ float red[256];
  int row = blockIdx.x;
  int t = threadIdx.x;
  float v = x[(size_t)row * DIN + t];
  red[t] = v; __syncthreads();
  for (int s = 128; s > 0; s >>= 1) { if (t < s) red[t] += red[t + s]; __syncthreads(); }
  float mean = red[0] * (1.0f / DIN);
  __syncthreads();
  float d = v - mean;
  red[t] = d * d; __syncthreads();
  for (int s = 128; s > 0; s >>= 1) { if (t < s) red[t] += red[t + s]; __syncthreads(); }
  float var = red[0] * (1.0f / DIN);
  float y = d * rsqrtf(var + 1e-5f) * g[t] + b[t];
  out[(size_t)row * DIN + t] = f2bf(y);
}

// ---------------- bf16 transpose (vb [NE][HD] -> vT [HD][NE]) ----------------
__global__ void transpose_bf16_kernel(const unsigned short* __restrict__ in,
                                      unsigned short* __restrict__ out) {
  __shared__ unsigned short tile[32][33];
  int c0 = blockIdx.x * 32;   // HD dim
  int r0 = blockIdx.y * 32;   // NE dim
  int tid = threadIdx.x;
  int tr = tid >> 5, tc = tid & 31;
  for (int rr = tr; rr < 32; rr += 8)
    tile[rr][tc] = in[(size_t)(r0 + rr) * HD + c0 + tc];
  __syncthreads();
  for (int rr = tr; rr < 32; rr += 8)
    out[(size_t)(c0 + rr) * NE + r0 + tc] = tile[tc][rr];
}

// ---------------- type-routed projection GEMM ----------------
// P = X_bf16[M x 256] @ Wsw[type][256 x 512]; store rows whose type == blockIdx.z.
// Block 128 (4 waves): tile 16(m) x 64(n). Async global->LDS staging, double-buffered.
__global__ void proj_kernel(const unsigned short* __restrict__ X, const unsigned short* __restrict__ W,
                            const int* __restrict__ types, unsigned short* __restrict__ out) {
  const int K = DIN, N = HD;
  int tt   = blockIdx.z;
  int row0 = blockIdx.y * 16;
  int col0 = blockIdx.x * 64;
  const unsigned short* Wsw = W + ((size_t)tt * (K >> 5) * (N >> 4) + (col0 >> 4)) * 512;
  __shared__ __align__(16) unsigned short As[2][512];
  __shared__ __align__(16) unsigned short Bs[2][2048];
  __shared__ __align__(16) unsigned short Ot[16 * 64];
  __shared__ int tysh[16];
  int tid = threadIdx.x, lane = tid & 31, wave = tid >> 5;
  if (tid < 16) tysh[tid] = types[row0 + tid];

  int am = tid & 15, alhi = (tid >> 4) & 1, ath = (tid >> 5) & 1;
  int ak0 = alhi * 16 + ath * 8;
  int adst = ((am + (alhi << 4)) << 4) + ath * 8;

  auto stage = [&](int kk, int buf) {
    if (tid < 64)
      async_ld_b128(&As[buf][adst], X + (size_t)(row0 + am) * K + kk + ak0);
    const unsigned short* bsrc = Wsw + (size_t)(kk >> 5) * ((N >> 4) * 512) + (tid << 4);
    async_ld_b128(&Bs[buf][tid << 4], bsrc);
    async_ld_b128(&Bs[buf][(tid << 4) + 8], bsrc + 8);
  };

  v8f acc = {};
  stage(0, 0);
  wait_async0();
  __syncthreads();
  const int NS = K / 32;
  for (int s = 0; s < NS; ++s) {
    if (s + 1 < NS) stage((s + 1) * 32, (s + 1) & 1);
    v16bf a = load_sw(As[s & 1], lane);
    v16bf b = load_sw(Bs[s & 1] + wave * 512, lane);
    acc = wmma_bf16(a, b, acc);
    wait_async0();
    __syncthreads();
  }
  int cn = lane & 15, mo = (lane < 16) ? 0 : 8;
#pragma unroll
  for (int r = 0; r < 8; ++r)
    Ot[(mo + r) * 64 + wave * 16 + cn] = f2bf(acc[r]);
  __syncthreads();
  int orow = tid >> 3, oseg = tid & 7;
  if (tysh[orow] == tt) {
    uint4 od = *(uint4*)&Ot[orow * 64 + oseg * 8];
    *(uint4*)&out[(size_t)(row0 + orow) * N + col0 + oseg * 8] = od;
  }
}

// ---------------- scores = q.k^T/8 + relpos, masked ----------------
// Block 128 (4 waves): tile 64(i) x 64(j) per head; async-staged panels, 8 WMMAs/wave;
// C staged to 16KB LDS then written as float4 with int4 incidence loads.
__global__ void scores_kernel(const unsigned short* __restrict__ Q, const unsigned short* __restrict__ Kb,
                              const int* __restrict__ inc, const float* __restrict__ rel,
                              float* __restrict__ Sc) {
  int h = blockIdx.z, i0 = blockIdx.y * 64, j0 = blockIdx.x * 64;
  int tid = threadIdx.x, lane = tid & 31, wave = tid >> 5;
  __shared__ __align__(16) unsigned short SH[16 * 512];   // Qs(8 tiles) + Ks(8 tiles); reused as 64x64 f32
  unsigned short* Qs = SH;
  unsigned short* Ks = SH + 8 * 512;
  __shared__ float rl[21];
  if (tid < 21) rl[tid] = rel[tid * NH + h];
#pragma unroll
  for (int gi = 0; gi < 4; ++gi) {   // Qs: 4 async b128 copies per thread
    int g = tid + gi * 128;
    int m = g & 63, gg = g >> 6;
    int kg = gg >> 2, lhi = gg & 1, th = (gg >> 1) & 1;
    int k0 = kg * 32 + lhi * 16 + th * 8;
    async_ld_b128(&Qs[(((m >> 4) * 2 + kg) << 9) + (((m & 15) + (lhi << 4)) << 4) + th * 8],
                  Q + (size_t)(i0 + m) * HD + h * DH + k0);
  }
#pragma unroll
  for (int gi = 0; gi < 2; ++gi) {   // Ks: 4 async b128 copies per thread
    int g = tid + gi * 128;          // 0..255
    int j = g & 63, seg = g >> 6;    // seg 0..3
    int kg = seg >> 1, lhi = seg & 1;
    int kd0 = kg * 32 + lhi * 16;
    const unsigned short* src = Kb + (size_t)(j0 + j) * HD + h * DH + kd0;
    unsigned short* dst = &Ks[((kg * 4 + (j >> 4)) << 9) + (((j & 15) + (lhi << 4)) << 4)];
    async_ld_b128(dst, src);
    async_ld_b128(dst + 8, src + 8);
  }
  wait_async0();
  __syncthreads();
  v8f acc[4] = {{}, {}, {}, {}};
#pragma unroll
  for (int kg = 0; kg < 2; ++kg) {
    v16bf a = load_sw(Qs + (wave * 2 + kg) * 512, lane);
#pragma unroll
    for (int jg = 0; jg < 4; ++jg) {
      v16bf b = load_sw(Ks + (kg * 4 + jg) * 512, lane);
      acc[jg] = wmma_bf16(a, b, acc[jg]);
    }
  }
  __syncthreads();
  float* Sblk = (float*)SH;          // 64 x 64 f32
  int cn = lane & 15, mo = (lane < 16) ? 0 : 8;
#pragma unroll
  for (int jg = 0; jg < 4; ++jg)
#pragma unroll
    for (int r = 0; r < 8; ++r)
      Sblk[(wave * 16 + mo + r) * 64 + jg * 16 + cn] = acc[jg][r];
  __syncthreads();
#pragma unroll
  for (int c = 0; c < 8; ++c) {
    int e = tid + c * 128;           // float4 id, 0..1023
    int ii = e >> 4, jj = (e & 15) * 4;
    int i = i0 + ii;
    int4 mrow = *(const int4*)&inc[(size_t)i * NE + j0 + jj];
    float4 sv = *(const float4*)&Sblk[ii * 64 + jj];
    const float so[4] = {sv.x, sv.y, sv.z, sv.w};
    const int*  mp = (const int*)&mrow;
    float4 res;
    float* rr = (float*)&res;
#pragma unroll
    for (int u = 0; u < 4; ++u) {
      int j = j0 + jj + u;
      float s = so[u] * 0.125f;      // 1/sqrt(64)
      int rp = i - j;
      rp = rp > MAXREL ? MAXREL : (rp < -MAXREL ? -MAXREL : rp);
      s += rl[rp + MAXREL];
      if (mp[u] == 0) s = -__builtin_inff();
      rr[u] = s;
    }
    *(float4*)&Sc[((size_t)h * NN + i) * NE + j0 + jj] = res;
  }
}

// ---------------- softmax row stats (float4 reads) ----------------
__global__ void softmax_stats_kernel(const float* __restrict__ Sc, float* __restrict__ rmax,
                                     float* __restrict__ rsum) {
  __shared__ float red[256];
  int i = blockIdx.x, h = blockIdx.y, t = threadIdx.x;
  const float* row = Sc + ((size_t)h * NN + i) * NE;
  float4 a = *(const float4*)&row[t * 4];
  float4 b = *(const float4*)&row[1024 + t * 4];
  float m = fmaxf(fmaxf(fmaxf(a.x, a.y), fmaxf(a.z, a.w)),
                  fmaxf(fmaxf(b.x, b.y), fmaxf(b.z, b.w)));
  red[t] = m; __syncthreads();
  for (int s = 128; s > 0; s >>= 1) { if (t < s) red[t] = fmaxf(red[t], red[t + s]); __syncthreads(); }
  float rm = red[0];
  __syncthreads();
  float sum = __expf(a.x - rm) + __expf(a.y - rm) + __expf(a.z - rm) + __expf(a.w - rm)
            + __expf(b.x - rm) + __expf(b.y - rm) + __expf(b.z - rm) + __expf(b.w - rm);
  red[t] = sum; __syncthreads();
  for (int s = 128; s > 0; s >>= 1) { if (t < s) red[t] += red[t + s]; __syncthreads(); }
  if (t == 0) { rmax[(size_t)h * NN + i] = rm; rsum[(size_t)h * NN + i] = red[0]; }
}

// ---------------- out = softmax(S) @ V (V transposed), SiLU -> bf16 ----------------
// Block 128 (4 waves): tile 32(i) x 64(d) per head, 2 accumulators/wave. Double-buffered;
// V panel staged by async DMA, weights computed on the fly.
__global__ void av_silu_kernel(const float* __restrict__ Sc, const float* __restrict__ rmax,
                               const float* __restrict__ rsum, const unsigned short* __restrict__ VT,
                               unsigned short* __restrict__ outs) {
  int h = blockIdx.y, i0 = blockIdx.x * 32;
  int tid = threadIdx.x, lane = tid & 31, wave = tid >> 5;
  __shared__ __align__(16) unsigned short Wts[2][1024];   // two A tiles (rows 0-15 / 16-31)
  __shared__ __align__(16) unsigned short Vs[2][2048];
  __shared__ __align__(16) unsigned short Ot[32 * 64];
  __shared__ float rmx[32], rsi[32];
  if (tid < 32) {
    rmx[tid] = rmax[(size_t)h * NN + i0 + tid];
    rsi[tid] = 1.0f / rsum[(size_t)h * NN + i0 + tid];
  }
  __syncthreads();
  const float* srow = Sc + ((size_t)h * NN + i0) * NE;
  const unsigned short* vh = VT + (size_t)h * DH * NE;

  int vd = tid & 63, vjh = tid >> 6;
  int vdst = ((vd >> 4) << 9) + (((vd & 15) + (vjh << 4)) << 4);

  auto stage = [&](int jb, int buf) {
    const unsigned short* vsrc = vh + (size_t)vd * NE + jb + vjh * 16;
    async_ld_b128(&Vs[buf][vdst], vsrc);
    async_ld_b128(&Vs[buf][vdst + 8], vsrc + 8);
#pragma unroll
    for (int gi = 0; gi < 2; ++gi) {   // W tiles: 32 rows x 8 j-quads = 256 groups
      int g = tid + gi * 128;
      int m = g >> 3, q = g & 7;
      int lhi = q >> 2, th = (q >> 1) & 1;
      float4 sv = *(const float4*)(srow + (size_t)m * NE + jb + q * 4);
      float e0 = __expf(sv.x - rmx[m]) * rsi[m];
      float e1 = __expf(sv.y - rmx[m]) * rsi[m];
      float e2 = __expf(sv.z - rmx[m]) * rsi[m];
      float e3 = __expf(sv.w - rmx[m]) * rsi[m];
      uint2 pk;
      pk.x = (unsigned int)f2bf(e0) | ((unsigned int)f2bf(e1) << 16);
      pk.y = (unsigned int)f2bf(e2) | ((unsigned int)f2bf(e3) << 16);
      *(uint2*)&Wts[buf][((m >> 4) << 9) + (((m & 15) + (lhi << 4)) << 4) + th * 8 + (q & 1) * 4] = pk;
    }
  };

  v8f acc0 = {}, acc1 = {};
  stage(0, 0);
  wait_async0();
  __syncthreads();
  const int NS = NE / 32;
  for (int s = 0; s < NS; ++s) {
    if (s + 1 < NS) stage((s + 1) * 32, (s + 1) & 1);
    v16bf b  = load_sw(Vs[s & 1] + wave * 512, lane);
    v16bf a0 = load_sw(Wts[s & 1], lane);
    v16bf a1 = load_sw(Wts[s & 1] + 512, lane);
    acc0 = wmma_bf16(a0, b, acc0);
    acc1 = wmma_bf16(a1, b, acc1);
    wait_async0();
    __syncthreads();
  }
  int cn = lane & 15, mo = (lane < 16) ? 0 : 8;
#pragma unroll
  for (int r = 0; r < 8; ++r) {
    float x0 = acc0[r];
    float x1 = acc1[r];
    Ot[(mo + r) * 64 + wave * 16 + cn]      = f2bf(x0 / (1.0f + __expf(-x0)));   // SiLU
    Ot[(16 + mo + r) * 64 + wave * 16 + cn] = f2bf(x1 / (1.0f + __expf(-x1)));
  }
  __syncthreads();
#pragma unroll
  for (int c = 0; c < 2; ++c) {
    int e = tid + c * 128;
    int orow = e >> 3, oseg = e & 7;
    uint4 od = *(uint4*)&Ot[orow * 64 + oseg * 8];
    *(uint4*)&outs[(size_t)(i0 + orow) * HD + h * DH + oseg * 8] = od;
  }
}

// ---------------- final = concat(n_ln, out) @ Wout + bout ----------------
// K = 768. Block 128 (4 waves): tile 16(m) x 64(n). Async staging; f32 LDS epilogue.
__global__ void final_gemm_kernel(const unsigned short* __restrict__ Nln,
                                  const unsigned short* __restrict__ Os,
                                  const unsigned short* __restrict__ Wb, const float* __restrict__ bout,
                                  float* __restrict__ out) {
  int col0 = blockIdx.x * 64;
  int row0 = blockIdx.y * 16;
  int tid = threadIdx.x, lane = tid & 31, wave = tid >> 5;
  __shared__ __align__(16) unsigned short As[2][512];
  __shared__ __align__(16) unsigned short Bs[2][2048];
  const unsigned short* Wsw = Wb + (size_t)(col0 >> 4) * 512;
  int am = tid & 15, alhi = (tid >> 4) & 1, ath = (tid >> 5) & 1;
  int ak0 = alhi * 16 + ath * 8;
  int adst = ((am + (alhi << 4)) << 4) + ath * 8;

  auto stage = [&](int kk, int buf) {
    if (tid < 64) {
      int kg = kk + ak0;   // 8-run never crosses the 256 concat boundary (kk%32==0)
      const unsigned short* sp = (kg < DIN) ? (Nln + (size_t)(row0 + am) * DIN + kg)
                                            : (Os  + (size_t)(row0 + am) * HD + (kg - DIN));
      async_ld_b128(&As[buf][adst], sp);
    }
    const unsigned short* bsrc = Wsw + (size_t)(kk >> 5) * ((DIN >> 4) * 512) + (tid << 4);
    async_ld_b128(&Bs[buf][tid << 4], bsrc);
    async_ld_b128(&Bs[buf][(tid << 4) + 8], bsrc + 8);
  };

  v8f acc = {};
  stage(0, 0);
  wait_async0();
  __syncthreads();
  const int NS = (DIN + HD) / 32;
  for (int s = 0; s < NS; ++s) {
    if (s + 1 < NS) stage((s + 1) * 32, (s + 1) & 1);
    v16bf a = load_sw(As[s & 1], lane);
    v16bf b = load_sw(Bs[s & 1] + wave * 512, lane);
    acc = wmma_bf16(a, b, acc);
    wait_async0();
    __syncthreads();
  }
  float* Ct = (float*)Bs;            // 16 x 64 f32 = 4KB (reuse)
  int cn = lane & 15, mo = (lane < 16) ? 0 : 8;
  float bo = bout[col0 + wave * 16 + cn];
#pragma unroll
  for (int r = 0; r < 8; ++r)
    Ct[(mo + r) * 64 + wave * 16 + cn] = acc[r] + bo;
  __syncthreads();
  int orow = tid >> 3, oseg = tid & 7;
  float4 f0 = *(float4*)&Ct[orow * 64 + oseg * 8];
  float4 f1 = *(float4*)&Ct[orow * 64 + oseg * 8 + 4];
  float4* op = (float4*)&out[(size_t)(row0 + orow) * DIN + col0 + oseg * 8];
  op[0] = f0; op[1] = f1;
}

// ---------------- Host launcher ----------------
extern "C" void kernel_launch(void* const* d_in, const int* in_sizes, int n_in,
                              void* d_out, int out_size, void* d_ws, size_t ws_size,
                              hipStream_t stream) {
  (void)in_sizes; (void)n_in; (void)out_size; (void)ws_size;
  const float* e_feat = (const float*)d_in[0];
  const float* n_feat = (const float*)d_in[1];
  const int*   inc    = (const int*)  d_in[2];
  const int*   ntype  = (const int*)  d_in[3];
  const int*   etype  = (const int*)  d_in[4];
  const float* Wq     = (const float*)d_in[5];
  const float* Wk     = (const float*)d_in[6];
  const float* Wv     = (const float*)d_in[7];
  const float* Wout   = (const float*)d_in[8];
  const float* bout   = (const float*)d_in[9];
  const float* ln_eg  = (const float*)d_in[10];
  const float* ln_eb  = (const float*)d_in[11];
  const float* ln_ng  = (const float*)d_in[12];
  const float* ln_nb  = (const float*)d_in[13];
  const float* rel    = (const float*)d_in[14];

  // workspace carve-up (~21 MB)
  char* ws = (char*)d_ws;
  unsigned short* n_ln = (unsigned short*)ws; ws += (size_t)NN * DIN * 2;
  unsigned short* e_ln = (unsigned short*)ws; ws += (size_t)NE * DIN * 2;
  unsigned short* qb   = (unsigned short*)ws; ws += (size_t)NN * HD * 2;
  unsigned short* kb   = (unsigned short*)ws; ws += (size_t)NE * HD * 2;
  unsigned short* vb   = (unsigned short*)ws; ws += (size_t)NE * HD * 2;
  unsigned short* vT   = (unsigned short*)ws; ws += (size_t)HD * NE * 2;
  unsigned short* osil = (unsigned short*)ws; ws += (size_t)NN * HD * 2;
  unsigned short* wqb  = (unsigned short*)ws; ws += (size_t)2 * DIN * HD * 2;
  unsigned short* wkb  = (unsigned short*)ws; ws += (size_t)4 * DIN * HD * 2;
  unsigned short* wvb  = (unsigned short*)ws; ws += (size_t)4 * DIN * HD * 2;
  unsigned short* wob  = (unsigned short*)ws; ws += (size_t)(DIN + HD) * DIN * 2;
  float* rmax = (float*)ws; ws += (size_t)NH * NN * 4;
  float* rsum = (float*)ws; ws += (size_t)NH * NN * 4;

  float* fout = (float*)d_out;                 // final [NN, DIN]
  float* Sc   = fout + (size_t)NN * DIN;       // scores [NH, NN, NE]

  // 0: weights -> bf16, pre-swizzled into WMMA B-tile layout (one pass)
  { int n8 = 2 * DIN * HD / 8; cvt_swz_kernel<<<(n8 + 255) / 256, 256, 0, stream>>>(Wq, wqb, DIN, HD, n8); }
  { int n8 = 4 * DIN * HD / 8; cvt_swz_kernel<<<(n8 + 255) / 256, 256, 0, stream>>>(Wk, wkb, DIN, HD, n8); }
  { int n8 = 4 * DIN * HD / 8; cvt_swz_kernel<<<(n8 + 255) / 256, 256, 0, stream>>>(Wv, wvb, DIN, HD, n8); }
  { int n8 = (DIN + HD) * DIN / 8; cvt_swz_kernel<<<(n8 + 255) / 256, 256, 0, stream>>>(Wout, wob, DIN + HD, DIN, n8); }

  // 1-2: LayerNorm -> bf16
  ln_bf16_kernel<<<NE, 256, 0, stream>>>(e_feat, ln_eg, ln_eb, e_ln);
  ln_bf16_kernel<<<NN, 256, 0, stream>>>(n_feat, ln_ng, ln_nb, n_ln);

  // 3: type-routed projections
  { dim3 g(HD / 64, NN / 16, 2); proj_kernel<<<g, 128, 0, stream>>>(n_ln, wqb, ntype, qb); }
  { dim3 g(HD / 64, NE / 16, 4); proj_kernel<<<g, 128, 0, stream>>>(e_ln, wkb, etype, kb); }
  { dim3 g(HD / 64, NE / 16, 4); proj_kernel<<<g, 128, 0, stream>>>(e_ln, wvb, etype, vb); }

  // 3b: V -> V^T for vectorized AV staging
  { dim3 g(HD / 32, NE / 32); transpose_bf16_kernel<<<g, 256, 0, stream>>>(vb, vT); }

  // 4: masked, biased scores straight into d_out's scores region
  { dim3 g(NE / 64, NN / 64, NH); scores_kernel<<<g, 128, 0, stream>>>(qb, kb, inc, rel, Sc); }

  // 5: softmax row stats
  { dim3 g(NN, NH); softmax_stats_kernel<<<g, 256, 0, stream>>>(Sc, rmax, rsum); }

  // 6: attention-weighted V + SiLU -> bf16
  { dim3 g(NN / 32, NH); av_silu_kernel<<<g, 128, 0, stream>>>(Sc, rmax, rsum, vT, osil); }

  // 7: final projection + bias -> d_out final region
  { dim3 g(DIN / 64, NN / 16); final_gemm_kernel<<<g, 128, 0, stream>>>(n_ln, osil, wob, bout, fout); }
}
